// RelGraph_29850022707649
// MI455X (gfx1250) — compile-verified, matrix operand order checked by reading, not run on hardware
//
#include <hip/hip_runtime.h>

// ---------------------------------------------------------------------------
// MI455X (gfx1250) wave32 implementation. All matmuls via
// v_wmma_f32_16x16x32_f16 (f16 inputs, fp32 accumulate); WMMA operand fetches
// are K-contiguous per lane -> ds_load_b128 pairs. Tile staging uses
// GLOBAL_LOAD_ASYNC_TO_LDS_B128 (ASYNCcnt, fire-and-forget) via inline asm,
// waited with s_wait_asynccnt before the barrier.
// ---------------------------------------------------------------------------

typedef __attribute__((ext_vector_type(16))) _Float16 v16h;
typedef __attribute__((ext_vector_type(8)))  float    v8f;

// Async global->LDS 16B copy. VDST = LDS byte address (wave-relative; the low
// 32 bits of a generic LDS pointer per ISA 10.2), VADDR = 64-bit global addr.
static __device__ __forceinline__ void async_cp16(void* lp, const void* gp) {
  asm volatile("global_load_async_to_lds_b128 %0, %1, off"
               :
               : "v"((unsigned)(unsigned long long)lp),
                 "v"((unsigned long long)gp)
               : "memory");
}
static __device__ __forceinline__ void wait_async() {
  asm volatile("s_wait_asynccnt 0x0" ::: "memory");
}

#define B_    128
#define K_    128
#define D_    768
#define H_    8
#define L_    4
#define HD_   96
#define FF_   3072
#define SIM_  192
#define BK_   (B_ * K_)
#define SCALE_ 0.1020620726159657f   // 96^-0.5
#define BOOST_ 0.3f
#define EPS_   1e-5f

static __device__ __forceinline__ v8f wmma32(v16h a, v16h b, v8f c) {
  return __builtin_amdgcn_wmma_f32_16x16x32_f16(false, a, false, b,
                                                (short)0, c, false, false);
}

// A-style fragment (16 rows x 32 K, f16, row-major, K contiguous per row).
// ISA 7.12.2: lane L -> row L&15; lane[4] selects K-half; elems 0..7 / 8..15
// are two contiguous 8-half (16B) runs -> compiles to 2x ds_load_b128.
static __device__ __forceinline__ v16h ldfragA(const _Float16* base, int row0,
                                               int kc, int stride) {
  const int l = threadIdx.x & 31;
  const _Float16* p =
      base + (size_t)(row0 + (l & 15)) * stride + kc + ((l & 16) ? 8 : 0);
  v16h r;
#pragma unroll
  for (int j = 0; j < 8; ++j) { r[j] = p[j]; r[8 + j] = p[16 + j]; }
  return r;
}

static __device__ __forceinline__ float gelu_exact(float v) {
  return 0.5f * v * (1.0f + erff(v * 0.7071067811865475f));
}

// ---------------------------------------------------------------------------
// Elementwise / normalization kernels
// ---------------------------------------------------------------------------

__global__ __launch_bounds__(256) void k_addpos(const float* __restrict__ slots,
                                                const float* __restrict__ pos,
                                                float* __restrict__ x) {
  const size_t i = (size_t)blockIdx.x * 256 + threadIdx.x;
  x[i] = slots[i] + pos[i % (K_ * D_)];
}

// Transposing f32 -> f16 weight conversion: out[l][n][k] = in[l][k][n].
__global__ __launch_bounds__(256) void k_convt(const float* __restrict__ in,
                                               _Float16* __restrict__ out,
                                               int Kd, int N) {
  __shared__ _Float16 tile[32][33];
  const int l = blockIdx.z;
  const int n0 = blockIdx.x * 32, k0 = blockIdx.y * 32;
  const int tx = threadIdx.x & 31, ty = threadIdx.x >> 5;  // 32 x 8
  const float* src = in + (size_t)l * Kd * N;
  _Float16* dst = out + (size_t)l * Kd * N;
#pragma unroll
  for (int r = 0; r < 4; ++r)
    tile[ty + r * 8][tx] = (_Float16)src[(size_t)(k0 + ty + r * 8) * N + n0 + tx];
  __syncthreads();
#pragma unroll
  for (int r = 0; r < 4; ++r)
    dst[(size_t)(n0 + ty + r * 8) * Kd + k0 + tx] = tile[tx][ty + r * 8];
}

// LayerNorm over D=768: one block/row, 3 elems/thread; f16 and/or f32 out.
__global__ __launch_bounds__(256) void k_ln(const float* __restrict__ x,
                                            const float* __restrict__ g,
                                            const float* __restrict__ bta,
                                            _Float16* __restrict__ outH,
                                            float* __restrict__ outF) {
  __shared__ float red[256];
  const int r = blockIdx.x, t = threadIdx.x;
  const float* row = x + (size_t)r * D_;
  const float v0 = row[t], v1 = row[t + 256], v2 = row[t + 512];
  red[t] = v0 + v1 + v2;
  __syncthreads();
  for (int o = 128; o > 0; o >>= 1) {
    if (t < o) red[t] += red[t + o];
    __syncthreads();
  }
  const float mean = red[0] * (1.0f / (float)D_);
  __syncthreads();
  const float d0 = v0 - mean, d1 = v1 - mean, d2 = v2 - mean;
  red[t] = d0 * d0 + d1 * d1 + d2 * d2;
  __syncthreads();
  for (int o = 128; o > 0; o >>= 1) {
    if (t < o) red[t] += red[t + o];
    __syncthreads();
  }
  const float inv = rsqrtf(red[0] * (1.0f / (float)D_) + EPS_);
  const float dv[3] = {d0, d1, d2};
#pragma unroll
  for (int j = 0; j < 3; ++j) {
    const int c = t + 256 * j;
    const float y = dv[j] * inv * g[c] + bta[c];
    if (outH) outH[(size_t)r * D_ + c] = (_Float16)y;
    if (outF) outF[(size_t)r * D_ + c] = y;
  }
}

// L2-normalize s rows (SIM=192) -> f16.
__global__ __launch_bounds__(192) void k_norm_s(const float* __restrict__ s,
                                                _Float16* __restrict__ s16) {
  __shared__ float red[192];
  const int r = blockIdx.x, t = threadIdx.x;
  const float v = s[(size_t)r * SIM_ + t];
  red[t] = v * v;
  __syncthreads();
  if (t < 96) red[t] += red[t + 96];
  __syncthreads();
  if (t < 48) red[t] += red[t + 48];
  __syncthreads();
  if (t < 24) red[t] += red[t + 24];
  __syncthreads();
  if (t < 12) red[t] += red[t + 12];
  __syncthreads();
  if (t < 6) red[t] += red[t + 6];
  __syncthreads();
  if (t < 3) red[t] += red[t + 3];
  __syncthreads();
  const float nrm = sqrtf(red[0] + red[1] + red[2]);
  const float inv = 1.0f / fmaxf(nrm, 1e-12f);
  s16[(size_t)r * SIM_ + t] = (_Float16)(v * inv);
}

// Row softmax over 128, one wave32 per row.
__global__ __launch_bounds__(256) void k_softmax(const float* __restrict__ sc,
                                                 _Float16* __restrict__ pr) {
  const int wave = threadIdx.x >> 5, lane = threadIdx.x & 31;
  const size_t row = (size_t)blockIdx.x * 8 + wave;
  const float* p = sc + row * K_;
  float v[4];
#pragma unroll
  for (int j = 0; j < 4; ++j) v[j] = p[lane + 32 * j];
  float m = fmaxf(fmaxf(v[0], v[1]), fmaxf(v[2], v[3]));
  for (int o = 16; o > 0; o >>= 1) m = fmaxf(m, __shfl_xor(m, o, 32));
  float e[4], s = 0.0f;
#pragma unroll
  for (int j = 0; j < 4; ++j) { e[j] = expf(v[j] - m); s += e[j]; }
  for (int o = 16; o > 0; o >>= 1) s += __shfl_xor(s, o, 32);
  const float inv = 1.0f / s;
  _Float16* q = pr + row * K_;
#pragma unroll
  for (int j = 0; j < 4; ++j) q[lane + 32 * j] = (_Float16)(e[j] * inv);
}

// ---------------------------------------------------------------------------
// WMMA GEMM: C[M,N] = A[M,K] * Bt[N,K]^T (+bias)(gelu)(+residual)
// Bt is the PRE-TRANSPOSED weight ([N][K]). Block = 256 threads, tile 256x64,
// Kc=32, double-buffered LDS. Tile kt+1 is staged with async global->LDS
// copies issued before the WMMAs of kt, waited just before the barrier.
// ---------------------------------------------------------------------------
#define TS 40  // LDS row stride in halves: 80B = 5*16B
__global__ __launch_bounds__(256, 1) void k_gemm(
    const _Float16* __restrict__ A, const _Float16* __restrict__ Bt,
    const float* __restrict__ bias, const float* __restrict__ res,
    float* __restrict__ outF, _Float16* __restrict__ outH, int M, int N,
    int Kd, int act) {
  __shared__ __align__(16) _Float16 As[2][256 * TS];
  __shared__ __align__(16) _Float16 Bs[2][64 * TS];
  const int t = threadIdx.x;
  const int bm = blockIdx.y, bn = blockIdx.x;
  const int wave = t >> 5;
  const int nk = Kd >> 5;

  auto stage = [&](int kt, int buf) {
#pragma unroll
    for (int p = 0; p < 4; ++p) {
      const int idx = t + (p << 8);
      async_cp16(&As[buf][(idx >> 2) * TS + ((idx & 3) << 3)],
                 A + (size_t)(bm * 256 + (idx >> 2)) * Kd + (kt << 5) +
                     ((idx & 3) << 3));
    }
    async_cp16(&Bs[buf][(t >> 2) * TS + ((t & 3) << 3)],
               Bt + (size_t)(bn * 64 + (t >> 2)) * Kd + (kt << 5) +
                   ((t & 3) << 3));
    if (kt + 1 < nk)  // global_prefetch_b8 one K-slice further ahead
      __builtin_prefetch(A + (size_t)(bm * 256 + t) * Kd + ((kt + 1) << 5), 0, 0);
  };

  v8f acc[2][4] = {};
  stage(0, 0);
  wait_async();
  __syncthreads();

  for (int kt = 0; kt < nk; ++kt) {
    const int cur = kt & 1;
    const bool more = (kt + 1 < nk);
    if (more) stage(kt + 1, cur ^ 1);  // async copies run under the WMMAs

    const _Float16* Ab = As[cur];
    const _Float16* Bb = Bs[cur];
    const v16h a0 = ldfragA(Ab, wave * 32, 0, TS);
    const v16h a1 = ldfragA(Ab, wave * 32 + 16, 0, TS);
#pragma unroll
    for (int j = 0; j < 4; ++j) {
      const v16h bfr = ldfragA(Bb, j * 16, 0, TS);
      acc[0][j] = wmma32(a0, bfr, acc[0][j]);
      acc[1][j] = wmma32(a1, bfr, acc[1][j]);
    }
    if (more) wait_async();
    __syncthreads();
  }

  const int l = t & 31;
  const int mo = (l & 16) ? 8 : 0;  // C/D layout: lane[4] -> M+8
  const int nin = l & 15;
#pragma unroll
  for (int i = 0; i < 2; ++i)
#pragma unroll
    for (int j = 0; j < 4; ++j) {
      const int row0 = bm * 256 + wave * 32 + i * 16;
      const int n = bn * 64 + j * 16 + nin;
      const float bi = bias ? bias[n] : 0.0f;
#pragma unroll
      for (int e = 0; e < 8; ++e) {
        const int m = row0 + mo + e;
        float v = acc[i][j][e] + bi;
        if (act) v = gelu_exact(v);
        if (res) v += res[(size_t)m * N + n];
        if (outF) outF[(size_t)m * N + n] = v;
        if (outH) outH[(size_t)m * N + n] = (_Float16)v;
      }
    }
}

// ---------------------------------------------------------------------------
// sim[b] = S_b (128x192) * S_b^T. Both fragments are A-pattern loads from the
// same row-major S tile (B-frag of a transpose == A-frag).
// ---------------------------------------------------------------------------
#define SS 200
__global__ __launch_bounds__(256, 1) void k_sim(const _Float16* __restrict__ s16,
                                                float* __restrict__ sim) {
  __shared__ __align__(16) _Float16 ss[128 * SS];
  const int b = blockIdx.x, t = threadIdx.x;
#pragma unroll
  for (int p2 = 0; p2 < 12; ++p2) {
    const int idx = t + (p2 << 8);
    const int row = idx / 24;
    const int c8  = (idx % 24) << 3;
    async_cp16(&ss[row * SS + c8], s16 + (size_t)(b * K_ + row) * SIM_ + c8);
  }
  wait_async();
  __syncthreads();
  const int m0 = (t >> 5) << 4;
  v16h af[6];
#pragma unroll
  for (int kt = 0; kt < 6; ++kt) af[kt] = ldfragA(ss, m0, kt << 5, SS);
  const int l = t & 31;
  const int mo = (l & 16) ? 8 : 0;
  const int nin = l & 15;
  for (int j = 0; j < 8; ++j) {
    v8f acc = {};
#pragma unroll
    for (int kt = 0; kt < 6; ++kt)
      acc = wmma32(af[kt], ldfragA(ss, j << 4, kt << 5, SS), acc);
    const int n = (j << 4) + nin;
#pragma unroll
    for (int e = 0; e < 8; ++e)
      sim[((size_t)b * K_ + (m0 + mo + e)) * K_ + n] = acc[e];
  }
}

// ---------------------------------------------------------------------------
// scores[b,h] = SCALE * q_bh * k_bh^T + BOOST*sim[b]. Block per (b,h).
// ---------------------------------------------------------------------------
#define QS 104
__global__ __launch_bounds__(256, 1) void k_scores(
    const _Float16* __restrict__ qh, const _Float16* __restrict__ kh,
    const float* __restrict__ sim, float* __restrict__ sc) {
  __shared__ __align__(16) _Float16 qs[128 * QS];
  __shared__ __align__(16) _Float16 ks[128 * QS];
  const int bh = blockIdx.x;
  const int b = bh / H_, h = bh % H_;
  const int t = threadIdx.x;
#pragma unroll
  for (int p2 = 0; p2 < 6; ++p2) {
    const int idx = t + (p2 << 8);
    const int row = idx / 12;
    const int c8  = (idx % 12) << 3;
    const size_t g = (size_t)(b * K_ + row) * D_ + h * HD_ + c8;
    async_cp16(&qs[row * QS + c8], qh + g);
    async_cp16(&ks[row * QS + c8], kh + g);
  }
  wait_async();
  __syncthreads();
  const int m0 = (t >> 5) << 4;
  v16h af[3];
#pragma unroll
  for (int kt = 0; kt < 3; ++kt) af[kt] = ldfragA(qs, m0, kt << 5, QS);
  const int l = t & 31;
  const int mo = (l & 16) ? 8 : 0;
  const int nin = l & 15;
  for (int j = 0; j < 8; ++j) {
    v8f acc = {};
#pragma unroll
    for (int kt = 0; kt < 3; ++kt)
      acc = wmma32(af[kt], ldfragA(ks, j << 4, kt << 5, QS), acc);  // k^T
    const int n = (j << 4) + nin;
#pragma unroll
    for (int e = 0; e < 8; ++e) {
      const int m = m0 + mo + e;
      sc[((size_t)bh * K_ + m) * K_ + n] =
          acc[e] * SCALE_ + BOOST_ * sim[((size_t)b * K_ + m) * K_ + n];
    }
  }
}

// ---------------------------------------------------------------------------
// o[b,h] = probs[b,h] (128x128) * v_bh (128x96). probs tile staged with async
// copies; v staged column-major (vs[n][k]) via paired-row b32 packing so its
// fragments are contiguous ds_load_b128 pairs.
// ---------------------------------------------------------------------------
#define PS  136
#define VSK 136
__global__ __launch_bounds__(256, 1) void k_av(const _Float16* __restrict__ pr,
                                               const _Float16* __restrict__ vh,
                                               _Float16* __restrict__ oh) {
  __shared__ __align__(16) _Float16 ps[128 * PS];
  __shared__ __align__(16) _Float16 vs[96 * VSK];  // vs[n][k] = v[k][n]
  const int bh = blockIdx.x;
  const int b = bh / H_, h = bh % H_;
  const int t = threadIdx.x;
#pragma unroll
  for (int p2 = 0; p2 < 8; ++p2) {
    const int idx = t + (p2 << 8);
    const int row = idx >> 4;
    const int c8  = (idx & 15) << 3;
    async_cp16(&ps[row * PS + c8], pr + ((size_t)bh * K_ + row) * K_ + c8);
  }
  // transpose-stage v: 64 k-pairs x 12 col-groups = 768 tasks, 3 per thread
#pragma unroll
  for (int p2 = 0; p2 < 3; ++p2) {
    const int task = t + (p2 << 8);
    const int kp = task / 12;
    const int cg = task % 12;
    const size_t g0 = (size_t)(b * K_ + 2 * kp) * D_ + h * HD_ + cg * 8;
    const uint4 r0 = *(const uint4*)(vh + g0);
    const uint4 r1 = *(const uint4*)(vh + g0 + D_);
    const unsigned short* h0 = (const unsigned short*)&r0;
    const unsigned short* h1 = (const unsigned short*)&r1;
#pragma unroll
    for (int jj = 0; jj < 8; ++jj) {
      const int n = cg * 8 + jj;
      *(unsigned int*)&vs[n * VSK + 2 * kp] =
          (unsigned int)h0[jj] | ((unsigned int)h1[jj] << 16);
    }
  }
  wait_async();
  __syncthreads();
  const int m0 = (t >> 5) << 4;
  v16h af[4];
#pragma unroll
  for (int kt = 0; kt < 4; ++kt) af[kt] = ldfragA(ps, m0, kt << 5, PS);
  const int l = t & 31;
  const int mo = (l & 16) ? 8 : 0;
  const int nin = l & 15;
  for (int j = 0; j < 6; ++j) {
    v8f acc = {};
#pragma unroll
    for (int kt = 0; kt < 4; ++kt)
      acc = wmma32(af[kt], ldfragA(vs, j << 4, kt << 5, VSK), acc);  // v^T stage
    const int n = (j << 4) + nin;
#pragma unroll
    for (int e = 0; e < 8; ++e)
      oh[(size_t)(b * K_ + (m0 + mo + e)) * D_ + h * HD_ + n] = (_Float16)acc[e];
  }
}

// ---------------------------------------------------------------------------
// Host orchestration
// ---------------------------------------------------------------------------
extern "C" void kernel_launch(void* const* d_in, const int* in_sizes, int n_in,
                              void* d_out, int out_size, void* d_ws,
                              size_t ws_size, hipStream_t stream) {
  (void)in_sizes; (void)n_in; (void)out_size; (void)ws_size;
  const float* slots = (const float*)d_in[0];
  const float* pos   = (const float*)d_in[1];
  const float* Wq    = (const float*)d_in[2];
  const float* Wk    = (const float*)d_in[3];
  const float* Wv    = (const float*)d_in[4];
  const float* Wo    = (const float*)d_in[5];
  const float* bo    = (const float*)d_in[6];
  const float* Wsim  = (const float*)d_in[7];
  const float* g1    = (const float*)d_in[8];
  const float* b1    = (const float*)d_in[9];
  const float* g2    = (const float*)d_in[10];
  const float* b2    = (const float*)d_in[11];
  const float* Wff1  = (const float*)d_in[12];
  const float* bff1  = (const float*)d_in[13];
  const float* Wff2  = (const float*)d_in[14];
  const float* bff2  = (const float*)d_in[15];
  const float* gf    = (const float*)d_in[16];
  const float* bf    = (const float*)d_in[17];

  char* wp = (char*)d_ws;
  auto alloc = [&](size_t bytes) -> void* {
    void* r = (void*)wp;
    wp += (bytes + 255) & ~(size_t)255;
    return r;
  };
  float*    xA = (float*)alloc((size_t)BK_ * D_ * 4);
  float*    xB = (float*)alloc((size_t)BK_ * D_ * 4);
  _Float16* nx = (_Float16*)alloc((size_t)BK_ * D_ * 2);
  _Float16* qh = (_Float16*)alloc((size_t)BK_ * D_ * 2);
  _Float16* kh = (_Float16*)alloc((size_t)BK_ * D_ * 2);
  _Float16* vh = (_Float16*)alloc((size_t)BK_ * D_ * 2);
  float*    sb = (float*)alloc((size_t)BK_ * SIM_ * 4);
  _Float16* sn = (_Float16*)alloc((size_t)BK_ * SIM_ * 2);
  float*    sm = (float*)alloc((size_t)B_ * K_ * K_ * 4);
  float*    sc = (float*)alloc((size_t)B_ * H_ * K_ * K_ * 4);
  _Float16* pr = (_Float16*)alloc((size_t)B_ * H_ * K_ * K_ * 2);
  _Float16* oh = (_Float16*)alloc((size_t)BK_ * D_ * 2);
  _Float16* hh = (_Float16*)alloc((size_t)BK_ * FF_ * 2);
  _Float16* Wq16   = (_Float16*)alloc((size_t)L_ * D_ * D_ * 2);    // [D][D]^T
  _Float16* Wk16   = (_Float16*)alloc((size_t)L_ * D_ * D_ * 2);
  _Float16* Wv16   = (_Float16*)alloc((size_t)L_ * D_ * D_ * 2);
  _Float16* Wo16   = (_Float16*)alloc((size_t)L_ * D_ * D_ * 2);
  _Float16* Wsim16 = (_Float16*)alloc((size_t)L_ * D_ * SIM_ * 2);  // [SIM][D]
  _Float16* Wff116 = (_Float16*)alloc((size_t)L_ * D_ * FF_ * 2);   // [FF][D]
  _Float16* Wff216 = (_Float16*)alloc((size_t)L_ * FF_ * D_ * 2);   // [D][FF]

  auto convt = [&](const float* src, _Float16* dst, int Kd, int N) {
    dim3 grid(N / 32, Kd / 32, L_);
    k_convt<<<grid, 256, 0, stream>>>(src, dst, Kd, N);
  };
  auto gemm = [&](const _Float16* A, const _Float16* Bt, const float* bias,
                  const float* res, float* oF, _Float16* oH, int M, int N,
                  int Kd, int act) {
    dim3 grid(N / 64, M / 256);
    k_gemm<<<grid, 256, 0, stream>>>(A, Bt, bias, res, oF, oH, M, N, Kd, act);
  };

  k_addpos<<<(B_ * K_ * D_) / 256, 256, 0, stream>>>(slots, pos, xA);

  convt(Wq, Wq16, D_, D_);
  convt(Wk, Wk16, D_, D_);
  convt(Wv, Wv16, D_, D_);
  convt(Wo, Wo16, D_, D_);
  convt(Wsim, Wsim16, D_, SIM_);
  convt(Wff1, Wff116, D_, FF_);
  convt(Wff2, Wff216, FF_, D_);

  for (int l = 0; l < L_; ++l) {
    const _Float16* Wql = Wq16 + (size_t)l * D_ * D_;
    const _Float16* Wkl = Wk16 + (size_t)l * D_ * D_;
    const _Float16* Wvl = Wv16 + (size_t)l * D_ * D_;
    const _Float16* Wol = Wo16 + (size_t)l * D_ * D_;
    const _Float16* Wsl = Wsim16 + (size_t)l * D_ * SIM_;
    const _Float16* W1l = Wff116 + (size_t)l * D_ * FF_;
    const _Float16* W2l = Wff216 + (size_t)l * FF_ * D_;

    // --- attention block (pre-norm) ---
    k_ln<<<BK_, 256, 0, stream>>>(xA, g1 + l * D_, b1 + l * D_, nx, nullptr);
    gemm(nx, Wql, nullptr, nullptr, nullptr, qh, BK_, D_, D_, 0);
    gemm(nx, Wkl, nullptr, nullptr, nullptr, kh, BK_, D_, D_, 0);
    gemm(nx, Wvl, nullptr, nullptr, nullptr, vh, BK_, D_, D_, 0);
    gemm(nx, Wsl, nullptr, nullptr, sb, nullptr, BK_, SIM_, D_, 0);
    k_norm_s<<<BK_, 192, 0, stream>>>(sb, sn);
    k_sim<<<B_, 256, 0, stream>>>(sn, sm);
    k_scores<<<B_ * H_, 256, 0, stream>>>(qh, kh, sm, sc);
    k_softmax<<<(B_ * H_ * K_) / 8, 256, 0, stream>>>(sc, pr);
    k_av<<<B_ * H_, 256, 0, stream>>>(pr, vh, oh);
    gemm(oh, Wol, bo + l * D_, xA, xB, nullptr, BK_, D_, D_, 0);

    // --- feed-forward block (pre-norm) ---
    k_ln<<<BK_, 256, 0, stream>>>(xB, g2 + l * D_, b2 + l * D_, nx, nullptr);
    gemm(nx, W1l, bff1 + l * FF_, nullptr, nullptr, hh, BK_, FF_, D_, 1);
    gemm(hh, W2l, bff2 + l * D_, xB, xA, nullptr, BK_, D_, FF_, 0);
  }

  k_ln<<<BK_, 256, 0, stream>>>(xA, gf, bf, nullptr, (float*)d_out);
}
